// ModernHopfield_91147795956075
// MI455X (gfx1250) — matrix operand order repaired and checked by reading, not compiled, not run on hardware
//
#include <hip/hip_runtime.h>

// Modern Hopfield fixed-point iteration on gfx1250 (MI455X).
// B=256, N=4096, M=16384, 50 steps. W converted once to bf16 (128 MB ->
// L2-resident), both GEMMs on v_wmma_f32_16x16x32_bf16 with fp32 accum.
// Global->LDS staging uses GLOBAL_LOAD_ASYNC_TO_LDS (ASYNCcnt) with
// double-buffered LDS; falls back to synchronous copies if the builtin is
// absent. Convergence freeze via device-side done flag (graph-safe).

#define B_DIM 256
#define N_DIM 4096
#define M_DIM 16384
#define NUM_STEPS 50

typedef __attribute__((ext_vector_type(16))) __bf16 v16bf;
typedef __attribute__((ext_vector_type(8)))  float  v8f;

typedef int v4i __attribute__((vector_size(16)));
typedef int v2i __attribute__((vector_size(8)));

union FragAB { v16bf v; unsigned u[8]; };

#if __has_builtin(__builtin_amdgcn_global_load_async_to_lds_b128)
#define USE_ASYNC 1
#else
#define USE_ASYNC 0
#endif

__device__ __forceinline__ void cp16_async(void* lds, const void* g) {
#if USE_ASYNC
  __builtin_amdgcn_global_load_async_to_lds_b128(
      (__attribute__((address_space(1))) v4i*)g,
      (__attribute__((address_space(3))) v4i*)lds, 0, 0);
#else
  *(uint4*)lds = *(const uint4*)g;
#endif
}

__device__ __forceinline__ void cp8_async(void* lds, const void* g) {
#if USE_ASYNC
  __builtin_amdgcn_global_load_async_to_lds_b64(
      (__attribute__((address_space(1))) v2i*)g,
      (__attribute__((address_space(3))) v2i*)lds, 0, 0);
#else
  *(uint2*)lds = *(const uint2*)g;
#endif
}

__device__ __forceinline__ void wait_async() {
#if USE_ASYNC
#if __has_builtin(__builtin_amdgcn_s_wait_asynccnt)
  __builtin_amdgcn_s_wait_asynccnt(0);
#else
  asm volatile("s_wait_asynccnt 0x0" ::: "memory");
#endif
#endif
}

__device__ __forceinline__ unsigned short f2bf(float f) {
  unsigned u = __builtin_bit_cast(unsigned, f);
  u += 0x7fffu + ((u >> 16) & 1u);          // round-to-nearest-even
  return (unsigned short)(u >> 16);
}

// ---------------- one-shot / control kernels ----------------

__global__ void k_convW(const float* __restrict__ W, unsigned short* __restrict__ Wb) {
  int i = blockIdx.x * blockDim.x + threadIdx.x;
  const int total = M_DIM * N_DIM;              // 67,108,864 < 2^31
  for (; i < total; i += gridDim.x * blockDim.x) Wb[i] = f2bf(W[i]);
}

__global__ void k_init_state(const float* __restrict__ in, float* __restrict__ st, int n) {
  int i = blockIdx.x * blockDim.x + threadIdx.x;
  for (; i < n; i += gridDim.x * blockDim.x) st[i] = in[i];
}

__global__ void k_reset(float* normPtr, int* donePtr) { *normPtr = 0.f; *donePtr = 0; }
__global__ void k_zero_norm(float* normPtr) { *normPtr = 0.f; }
__global__ void k_check(const float* normPtr, int* donePtr) {
  if (!*donePtr && *normPtr <= 1e-6f) *donePtr = 1;   // upd = sqrt(sum) <= 1e-3
}

// state[b][n] fp32 -> Sb[b][n] bf16 (straight cast; layout is already K-major
// per column b for the GEMM1 B-matrix)
__global__ void k_prep_state(const float* __restrict__ state,
                             unsigned short* __restrict__ Sb,
                             const int* __restrict__ done) {
  if (*done) return;
  int i = blockIdx.x * blockDim.x + threadIdx.x;
  const int total = B_DIM * N_DIM;
  for (; i < total; i += gridDim.x * blockDim.x) Sb[i] = f2bf(state[i]);
}

// ---------------- GEMM1: h[b][m] = beta * sum_n W[m][n]*state[b][n] ----------------
// C tile 128(M) x 64(b cols) per WG, 8 waves (4x2), wave tile 32x32 (2x2 frags).
// LDS double-buffered; staged by async global->LDS b128 copies.

__global__ __launch_bounds__(256)
void k_gemm_h(const unsigned short* __restrict__ Wb,
              const unsigned short* __restrict__ Sb,
              float* __restrict__ hbuf,
              const float* __restrict__ betaPtr,
              const int* __restrict__ done) {
  if (*done) return;
  __shared__ __align__(16) unsigned short lsA[2][128 * 40];  // [row m][k]
  __shared__ __align__(16) unsigned short lsB[2][64 * 40];   // [col b][k]

  const int tid = threadIdx.x;
  const int m0 = blockIdx.x * 128;
  const int b0 = blockIdx.y * 64;
  const int wid = tid >> 5, lane = tid & 31;
  const int waveM = wid >> 1, waveN = wid & 1;
  const int hi = lane >> 4, l15 = lane & 15;

  v8f c[2][2] = {};

  const int ar = tid >> 1;            // A stage: row 0..127
  const int ac = (tid & 1) * 16;      //          col 0/16 (2x b128)
  const int bcol = tid >> 2;          // B stage: col 0..63
  const int bks = (tid & 3) * 8;      //          k 0..24  (1x b128)

  const unsigned short* aSrc = Wb + (m0 + ar) * N_DIM + ac;
  const unsigned short* bSrc = Sb + (b0 + bcol) * N_DIM + bks;

  // prologue: stage tile 0 into buffer 0
  cp16_async(&lsA[0][ar * 40 + ac],     aSrc);
  cp16_async(&lsA[0][ar * 40 + ac + 8], aSrc + 8);
  cp16_async(&lsB[0][bcol * 40 + bks],  bSrc);

  int buf = 0;
  for (int k0 = 0; k0 < N_DIM; k0 += 32) {
    wait_async();
    __syncthreads();
    if (k0 + 32 < N_DIM) {            // prefetch next tile into other buffer
      const int nb = buf ^ 1, kn = k0 + 32;
      cp16_async(&lsA[nb][ar * 40 + ac],     aSrc + kn);
      cp16_async(&lsA[nb][ar * 40 + ac + 8], aSrc + kn + 8);
      cp16_async(&lsB[nb][bcol * 40 + bks],  bSrc + kn);
    }

    FragAB a[2], b[2];
#pragma unroll
    for (int mi = 0; mi < 2; ++mi) {          // A 16x32 bf16 layout
      const int row = waveM * 32 + mi * 16 + l15;
#pragma unroll
      for (int v = 0; v < 8; ++v) {
        const int kb = (v < 4 ? 2 * v : 2 * v + 8) + (hi ? 8 : 0);
        a[mi].u[v] = *(const unsigned*)&lsA[buf][row * 40 + kb];
      }
    }
#pragma unroll
    for (int ni = 0; ni < 2; ++ni) {          // B 32x16 bf16 layout
      const int col = waveN * 32 + ni * 16 + l15;
#pragma unroll
      for (int v = 0; v < 8; ++v) {
        const int kb = (v < 4 ? 2 * v : 2 * v + 8) + (hi ? 8 : 0);
        b[ni].u[v] = *(const unsigned*)&lsB[buf][col * 40 + kb];
      }
    }
#pragma unroll
    for (int mi = 0; mi < 2; ++mi)
#pragma unroll
      for (int ni = 0; ni < 2; ++ni)
        c[mi][ni] = __builtin_amdgcn_wmma_f32_16x16x32_bf16(
            false, a[mi].v, false, b[ni].v, (short)0, c[mi][ni], false, false);
    buf ^= 1;
  }

  const float beta = *betaPtr;
#pragma unroll
  for (int mi = 0; mi < 2; ++mi)
#pragma unroll
    for (int ni = 0; ni < 2; ++ni) {
      const int bc = b0 + waveN * 32 + ni * 16 + l15;
      const int mbase = m0 + waveM * 32 + mi * 16 + hi * 8;
#pragma unroll
      for (int r = 0; r < 8; ++r)             // transposed store -> h[b][m]
        hbuf[bc * M_DIM + mbase + r] = beta * c[mi][ni][r];
    }
}

// ---------------- softmax over M per batch row: f^T[b][m] (bf16) ----------------

__global__ __launch_bounds__(256)
void k_softmax(const float* __restrict__ hbuf,
               unsigned short* __restrict__ fT,
               const int* __restrict__ done) {
  if (*done) return;
  __shared__ float red[256];
  const int b = blockIdx.x;
  const int tid = threadIdx.x;
  const float* row = hbuf + b * M_DIM;

  float mx = -3.402823466e38f;
  for (int m = tid; m < M_DIM; m += 256) mx = fmaxf(mx, row[m]);
  red[tid] = mx; __syncthreads();
  for (int s = 128; s > 0; s >>= 1) {
    if (tid < s) red[tid] = fmaxf(red[tid], red[tid + s]);
    __syncthreads();
  }
  mx = red[0]; __syncthreads();

  float sum = 0.f;
  for (int m = tid; m < M_DIM; m += 256) sum += __expf(row[m] - mx);
  red[tid] = sum; __syncthreads();
  for (int s = 128; s > 0; s >>= 1) {
    if (tid < s) red[tid] += red[tid + s];
    __syncthreads();
  }
  const float inv = 1.0f / red[0];

  unsigned short* out = fT + b * M_DIM;
  for (int m = tid; m < M_DIM; m += 256) out[m] = f2bf(__expf(row[m] - mx) * inv);
}

// ---------------- GEMM2: v^T[b][n] = f^T[b][:] @ W[:,n]; in-place update ----------
// C tile 32(b rows) x 128(n cols) per WG, 8 waves (1x8), wave tile 32x16.
// B tile staged K-major ([k][n], async b128); per-wave B fragment gathered as
// u16 pairs (K-contiguous per column comes from strided LDS rows).

__global__ __launch_bounds__(256)
void k_gemm_v(const unsigned short* __restrict__ fT,
              const unsigned short* __restrict__ Wb,
              float* __restrict__ state,
              const float* __restrict__ tauPtr,
              float* __restrict__ normPtr,
              const int* __restrict__ done) {
  if (*done) return;
  __shared__ __align__(16) unsigned short lsA[2][32 * 40];   // [row b][k=m]
  __shared__ __align__(16) unsigned short lsB[2][32 * 136];  // [k=m][col n]

  const int tid = threadIdx.x;
  const int b0 = blockIdx.x * 32;
  const int n0 = blockIdx.y * 128;
  const int wid = tid >> 5, lane = tid & 31;
  const int hi = lane >> 4, l15 = lane & 15;

  v8f c[2] = {};

  const int ar = tid >> 3;            // A stage: row 0..31
  const int ac = (tid & 7) * 4;       //          col 0..28 (1x b64)
  const int bkr = tid >> 3;           // B stage: k-row 0..31
  const int bc0 = (tid & 7) * 16;     //          col 0..112 (2x b128)

  const unsigned short* aSrc = fT + (b0 + ar) * M_DIM + ac;
  const unsigned short* bSrc = Wb + bkr * N_DIM + n0 + bc0;

  cp8_async (&lsA[0][ar * 40 + ac],        aSrc);
  cp16_async(&lsB[0][bkr * 136 + bc0],     bSrc);
  cp16_async(&lsB[0][bkr * 136 + bc0 + 8], bSrc + 8);

  int buf = 0;
  for (int k0 = 0; k0 < M_DIM; k0 += 32) {
    wait_async();
    __syncthreads();
    if (k0 + 32 < M_DIM) {
      const int nb = buf ^ 1, kn = k0 + 32;
      cp8_async (&lsA[nb][ar * 40 + ac],        aSrc + kn);
      cp16_async(&lsB[nb][bkr * 136 + bc0],     bSrc + (size_t)kn * N_DIM);
      cp16_async(&lsB[nb][bkr * 136 + bc0 + 8], bSrc + (size_t)kn * N_DIM + 8);
    }

    FragAB a[2], bb;
#pragma unroll
    for (int mi = 0; mi < 2; ++mi) {
      const int row = mi * 16 + l15;
#pragma unroll
      for (int v = 0; v < 8; ++v) {
        const int kb = (v < 4 ? 2 * v : 2 * v + 8) + (hi ? 8 : 0);
        a[mi].u[v] = *(const unsigned*)&lsA[buf][row * 40 + kb];
      }
    }
    {
      const int col = wid * 16 + l15;
#pragma unroll
      for (int v = 0; v < 8; ++v) {
        const int k = (v < 4 ? 2 * v : 2 * v + 8) + (hi ? 8 : 0);
        const unsigned lo = lsB[buf][k * 136 + col];
        const unsigned h2 = lsB[buf][(k + 1) * 136 + col];
        bb.u[v] = lo | (h2 << 16);
      }
    }
#pragma unroll
    for (int mi = 0; mi < 2; ++mi)
      c[mi] = __builtin_amdgcn_wmma_f32_16x16x32_bf16(
          false, a[mi].v, false, bb.v, (short)0, c[mi], false, false);
    buf ^= 1;
  }

  // Epilogue: new = s + (v - s)/tau (dt=1); accumulate ||new - s||^2 globally.
  const float invTau = 1.0f / *tauPtr;
  float acc = 0.f;
  const int n = n0 + wid * 16 + l15;
#pragma unroll
  for (int mi = 0; mi < 2; ++mi) {
    const int bbase = b0 + mi * 16 + hi * 8;
#pragma unroll
    for (int r = 0; r < 8; ++r) {
      const int bidx = bbase + r;
      const float s = state[bidx * N_DIM + n];
      const float d = (c[mi][r] - s) * invTau;
      state[bidx * N_DIM + n] = s + d;
      acc += d * d;
    }
  }
#pragma unroll
  for (int m = 16; m >= 1; m >>= 1) acc += __shfl_xor(acc, m, 32);
  if (lane == 0) atomicAdd(normPtr, acc);
}

// ---------------- host driver ----------------

extern "C" void kernel_launch(void* const* d_in, const int* in_sizes, int n_in,
                              void* d_out, int out_size, void* d_ws, size_t ws_size,
                              hipStream_t stream) {
  const float* inp  = (const float*)d_in[0];   // [B,N,1]
  const float* W    = (const float*)d_in[1];   // [M,N]
  const float* beta = (const float*)d_in[2];   // scalar
  const float* tau  = (const float*)d_in[3];   // scalar
  float* state = (float*)d_out;                // live state, [B][N]

  char* ws = (char*)d_ws;
  size_t off = 0;
  unsigned short* Wb = (unsigned short*)(ws + off); off += (size_t)M_DIM * N_DIM * 2; // 128 MB
  unsigned short* Sb = (unsigned short*)(ws + off); off += (size_t)B_DIM * N_DIM * 2; // 2 MB
  float* hbuf        = (float*)(ws + off);          off += (size_t)B_DIM * M_DIM * 4; // 16 MB
  unsigned short* fT = (unsigned short*)(ws + off); off += (size_t)B_DIM * M_DIM * 2; // 8 MB
  float* normPtr     = (float*)(ws + off);
  int*   donePtr     = (int*)(ws + off + 4);
  (void)ws_size; (void)in_sizes; (void)n_in; (void)out_size;

  k_convW<<<4096, 256, 0, stream>>>(W, Wb);                    // once per call
  k_init_state<<<1024, 256, 0, stream>>>(inp, state, B_DIM * N_DIM);
  k_reset<<<1, 1, 0, stream>>>(normPtr, donePtr);

  for (int step = 0; step < NUM_STEPS; ++step) {
    k_zero_norm<<<1, 1, 0, stream>>>(normPtr);
    k_prep_state<<<1024, 256, 0, stream>>>(state, Sb, donePtr);
    k_gemm_h<<<dim3(M_DIM / 128, B_DIM / 64), 256, 0, stream>>>(Wb, Sb, hbuf, beta, donePtr);
    k_softmax<<<B_DIM, 256, 0, stream>>>(hbuf, fT, donePtr);
    k_gemm_v<<<dim3(B_DIM / 32, N_DIM / 128), 256, 0, stream>>>(fT, Wb, state, tau, normPtr, donePtr);
    k_check<<<1, 1, 0, stream>>>(normPtr, donePtr);
  }
}
